// GroupSpecificHeads_62723702391681
// MI455X (gfx1250) — matrix operand order repaired and checked by reading, not compiled, not run on hardware
//
#include <hip/hip_runtime.h>
#include <hip/hip_bf16.h>
#include <math.h>

typedef __attribute__((ext_vector_type(16))) _Float16 v16h;
typedef __attribute__((ext_vector_type(8)))  _Float16 v8h;
typedef __attribute__((ext_vector_type(8)))  float    v8f;
typedef __attribute__((ext_vector_type(4)))  float    v4f;

#define WPB 8   // waves per block in the MLP kernel

__device__ __forceinline__ float gelu_exact(float x) {
    return 0.5f * x * (1.0f + erff(x * 0.70710678118654752f));
}

// ---------------- Kernel 1: per-block group histogram ----------------
__global__ __launch_bounds__(256) void count_kernel(const int* __restrict__ groups,
                                                    int* __restrict__ blkcnt, int E, int G) {
    __shared__ int cnt[8];
    if (threadIdx.x < 8) cnt[threadIdx.x] = 0;
    __syncthreads();
    int i = blockIdx.x * 256 + threadIdx.x;
    if (i < E) atomicAdd(&cnt[groups[i]], 1);
    __syncthreads();
    if ((int)threadIdx.x < G) blkcnt[blockIdx.x * G + threadIdx.x] = cnt[threadIdx.x];
}

// ---------------- Kernel 2: sequential scan (tiny: nblocks*G) ----------------
__global__ void scan_kernel(const int* __restrict__ blkcnt, int* __restrict__ blkoff,
                            int* __restrict__ meta, int nblocks, int G) {
    if (threadIdx.x != 0 || blockIdx.x != 0) return;
    int totals[8];
    for (int g = 0; g < G; ++g) totals[g] = 0;
    for (int b = 0; b < nblocks; ++b)
        for (int g = 0; g < G; ++g) {
            blkoff[b * G + g] = totals[g];
            totals[g] += blkcnt[b * G + g];
        }
    int gs[9], tb[9];
    gs[0] = 0; tb[0] = 0;
    for (int g = 0; g < G; ++g) {
        gs[g + 1] = gs[g] + totals[g];
        tb[g + 1] = tb[g] + (totals[g] + 15) / 16;
    }
    for (int b = 0; b < nblocks; ++b)
        for (int g = 0; g < G; ++g) blkoff[b * G + g] += gs[g];
    meta[0] = tb[G];                       // total tiles
    for (int g = 0; g <= G; ++g) { meta[1 + g] = tb[g]; meta[10 + g] = gs[g]; }
}

// ---------------- Kernel 3: stable counting-sort scatter ----------------
__global__ __launch_bounds__(256) void scatter_kernel(const int* __restrict__ groups,
                                                      const int* __restrict__ blkoff,
                                                      int* __restrict__ sorted, int E, int G) {
    __shared__ int wcnt[8][8];
    int lane = threadIdx.x & 31;
    int wid  = threadIdx.x >> 5;
    int i = blockIdx.x * 256 + threadIdx.x;
    int g = (i < E) ? groups[i] : -1;
    int rank = 0;
    unsigned long long lt = ((unsigned long long)1 << lane) - 1ull;
    for (int gg = 0; gg < G; ++gg) {
        unsigned long long m = __ballot(g == gg);
        if (g == gg) rank = __popcll(m & lt);
        if (lane == 0) wcnt[wid][gg] = __popcll(m);
    }
    __syncthreads();
    if (g >= 0) {
        int off = 0;
        for (int w = 0; w < wid; ++w) off += wcnt[w][g];
        sorted[blkoff[blockIdx.x * G + g] + off + rank] = i;
    }
}

// ---------------- Kernel 4: weight transpose + f32->f16 ----------------
// W1t[g][n(128)][k(256)] = W1[g][k][n] ; W2t[g][n(64)][k(128)] = W2[g][k][n]
__global__ __launch_bounds__(256) void prep_weights(const float* __restrict__ W1,
                                                    const float* __restrict__ W2,
                                                    _Float16* __restrict__ W1t,
                                                    _Float16* __restrict__ W2t, int G) {
    int n1 = G * 128 * 256;
    int n2 = G * 64 * 128;
    int i = blockIdx.x * 256 + threadIdx.x;
    if (i < n1) {
        int d = i & 255;               // k  (0..255)
        int h = (i >> 8) & 127;        // n  (0..127)
        int g = i >> 15;
        W1t[i] = (_Float16)W1[((size_t)g * 256 + d) * 128 + h];
    } else if (i < n1 + n2) {
        int j = i - n1;
        int d = j & 127;               // k  (0..127)
        int n = (j >> 7) & 63;         // n  (0..63)
        int g = j >> 13;
        W2t[j] = (_Float16)W2[((size_t)g * 128 + d) * 64 + n];
    }
}

// ---------------- Kernel 5: fused group-head MLP, one wave per 16-edge tile ----------------
// Dynamic LDS arena: hbuf (WPB*16*128 f16 = 32KB) | w1s (128*256 f16 = 64KB) | w2s (64*128 f16 = 16KB)
#define LDS_HBUF_HALFS (WPB * 16 * 128)
#define LDS_W1_HALFS   (128 * 256)
#define LDS_W2_HALFS   (64 * 128)

__global__ __launch_bounds__(256) void mlp_kernel(
    const float* __restrict__ emb, const int* __restrict__ edges,
    const float* __restrict__ b1, const float* __restrict__ lng, const float* __restrict__ lnb,
    const float* __restrict__ b2, const float* __restrict__ W3, const float* __restrict__ b3,
    const _Float16* __restrict__ W1t, const _Float16* __restrict__ W2t,
    const int* __restrict__ sorted, const int* __restrict__ meta,
    float* __restrict__ out, int G) {

    extern __shared__ _Float16 smem[];
    _Float16* hbufA = smem;                         // per-wave activation staging
    _Float16* w1s   = smem + LDS_HBUF_HALFS;        // block's primary-group W1t
    _Float16* w2s   = w1s + LDS_W1_HALFS;           // block's primary-group W2t

    const int lane = threadIdx.x & 31;
    const int wid  = threadIdx.x >> 5;
    const int tile = blockIdx.x * WPB + wid;
    const int total_tiles = meta[0];

    // ---- stage the block's primary group weights into LDS (8x traffic amortization) ----
    const int tile0 = blockIdx.x * WPB;
    int gp = 0;
    if (tile0 < total_tiles) {
        while (gp + 1 < G && tile0 >= meta[1 + gp + 1]) ++gp;
        {
            const v8h* s1 = (const v8h*)(W1t + (size_t)gp * LDS_W1_HALFS);
            v8h*       d1 = (v8h*)w1s;
#pragma unroll
            for (int it = 0; it < LDS_W1_HALFS / 8 / 256; ++it)
                d1[it * 256 + threadIdx.x] = s1[it * 256 + threadIdx.x];
            const v8h* s2 = (const v8h*)(W2t + (size_t)gp * LDS_W2_HALFS);
            v8h*       d2 = (v8h*)w2s;
#pragma unroll
            for (int it = 0; it < LDS_W2_HALFS / 8 / 256; ++it)
                d2[it * 256 + threadIdx.x] = s2[it * 256 + threadIdx.x];
        }
    }
    __syncthreads();
    if (tile >= total_tiles) return;

    // find this tile's group via tile_base[]
    int g = 0;
    while (g + 1 < G && tile >= meta[1 + g + 1]) ++g;
    const int tl     = tile - meta[1 + g];
    const int gstart = meta[10 + g];
    const int gend   = meta[10 + g + 1];

    // B-weight base pointers: LDS (common case, g == gp) or global (block-boundary waves).
    // Generic (flat) pointers give one code path for both.
    const _Float16* w1base = (g == gp) ? (const _Float16*)w1s
                                       : (W1t + (size_t)g * LDS_W1_HALFS);
    const _Float16* w2base = (g == gp) ? (const _Float16*)w2s
                                       : (W2t + (size_t)g * LDS_W2_HALFS);

    const int row16 = lane & 15;     // A-row / B-col index within the 16-wide tile
    const int hi16  = lane >> 4;     // half-wave selector (K sub-interleave)
    const int koff  = hi16 * 8;      // A lane K offset

    // my row's edge (clamped for padding rows; padded results are discarded)
    const int prow = gstart + tl * 16 + row16;
    const int pc   = (prow < gend) ? prow : (gend - 1);
    const int e    = sorted[pc];
    const int u    = edges[2 * e];
    const int v    = edges[2 * e + 1];
    const float* rowu = emb + (size_t)u * 128;
    const float* rowv = emb + (size_t)v * 128;

    // ---------------- Layer 1: [16x256] x [256x128] via 8x8 WMMA(16x16x32 f16) ----------------
    v8f acc[8];
#pragma unroll
    for (int nt = 0; nt < 8; ++nt) acc[nt] = (v8f){0.f,0.f,0.f,0.f,0.f,0.f,0.f,0.f};

#pragma unroll
    for (int kt = 0; kt < 8; ++kt) {
        // A fragment: a[0..7] = ee[row][kt*32+koff+t], a[8..15] = ee[row][kt*32+16+koff+t]
        const int c0 = kt * 32 + koff;
        const int c1 = c0 + 16;
        const float* p0 = (c0 < 128) ? (rowu + c0) : (rowv + (c0 - 128));
        const float* p1 = (c1 < 128) ? (rowu + c1) : (rowv + (c1 - 128));
        v4f f0 = *(const v4f*)p0, f1 = *(const v4f*)(p0 + 4);
        v4f f2 = *(const v4f*)p1, f3 = *(const v4f*)(p1 + 4);
        v16h a;
#pragma unroll
        for (int t = 0; t < 4; ++t) {
            a[t]      = (_Float16)f0[t];
            a[4 + t]  = (_Float16)f1[t];
            a[8 + t]  = (_Float16)f2[t];
            a[12 + t] = (_Float16)f3[t];
        }
#pragma unroll
        for (int nt = 0; nt < 8; ++nt) {
            // B fragment: col n = nt*16+row16, b[0..7] = K kt*32+hi16*8+t, b[8..15] = K kt*32+16+hi16*8+t
            const _Float16* wp = w1base + ((size_t)(nt * 16 + row16)) * 256 + kt * 32 + hi16 * 8;
            v8h b0  = *(const v8h*)wp;
            v8h b1h = *(const v8h*)(wp + 16);
            v16h b;
#pragma unroll
            for (int t = 0; t < 8; ++t) { b[t] = b0[t]; b[8 + t] = b1h[t]; }
            acc[nt] = __builtin_amdgcn_wmma_f32_16x16x32_f16(false, a, false, b,
                                                             (short)0, acc[nt], false, false);
        }
    }

    // ---------------- bias + LayerNorm (row = VGPR j (+8 for upper half), col = nt*16+row16) ----
#pragma unroll
    for (int nt = 0; nt < 8; ++nt) {
        float bb = b1[g * 128 + nt * 16 + row16];
#pragma unroll
        for (int j = 0; j < 8; ++j) acc[nt][j] += bb;
    }
    float s1v[8], s2v[8];
#pragma unroll
    for (int j = 0; j < 8; ++j) { s1v[j] = 0.f; s2v[j] = 0.f; }
#pragma unroll
    for (int nt = 0; nt < 8; ++nt)
#pragma unroll
        for (int j = 0; j < 8; ++j) { float x = acc[nt][j]; s1v[j] += x; s2v[j] += x * x; }
#pragma unroll
    for (int m = 1; m < 16; m <<= 1)
#pragma unroll
        for (int j = 0; j < 8; ++j) {
            s1v[j] += __shfl_xor(s1v[j], m, 32);
            s2v[j] += __shfl_xor(s2v[j], m, 32);
        }
    float mu[8], rstd[8];
#pragma unroll
    for (int j = 0; j < 8; ++j) {
        mu[j] = s1v[j] * (1.0f / 128.0f);
        float var = s2v[j] * (1.0f / 128.0f) - mu[j] * mu[j];
        rstd[j] = rsqrtf(var + 1e-5f);
    }
    // normalize + GELU -> f16 LDS (row-major 16x128) for the layer-2 A transpose
    _Float16* hb = hbufA + wid * (16 * 128);
#pragma unroll
    for (int nt = 0; nt < 8; ++nt) {
        const int col = nt * 16 + row16;
        const float gam = lng[g * 128 + col];
        const float bet = lnb[g * 128 + col];
#pragma unroll
        for (int j = 0; j < 8; ++j) {
            float x = (acc[nt][j] - mu[j]) * rstd[j] * gam + bet;
            x = gelu_exact(x);
            const int r = j + hi16 * 8;
            hb[r * 128 + col] = (_Float16)x;
        }
    }
    asm volatile("s_wait_dscnt 0x0" ::: "memory");   // per-wave LDS write->read fence

    // ---------------- Layer 2: [16x128] x [128x64] via 4x4 WMMA ----------------
    v8f acc2[4];
#pragma unroll
    for (int nt = 0; nt < 4; ++nt) acc2[nt] = (v8f){0.f,0.f,0.f,0.f,0.f,0.f,0.f,0.f};
#pragma unroll
    for (int kt = 0; kt < 4; ++kt) {
        const _Float16* ap = hb + row16 * 128 + kt * 32 + koff;
        v8h a0 = *(const v8h*)ap;
        v8h a1 = *(const v8h*)(ap + 16);
        v16h a;
#pragma unroll
        for (int t = 0; t < 8; ++t) { a[t] = a0[t]; a[8 + t] = a1[t]; }
#pragma unroll
        for (int nt = 0; nt < 4; ++nt) {
            const _Float16* wp = w2base + ((size_t)(nt * 16 + row16)) * 128 + kt * 32 + hi16 * 8;
            v8h b0  = *(const v8h*)wp;
            v8h b1h = *(const v8h*)(wp + 16);
            v16h b;
#pragma unroll
            for (int t = 0; t < 8; ++t) { b[t] = b0[t]; b[8 + t] = b1h[t]; }
            acc2[nt] = __builtin_amdgcn_wmma_f32_16x16x32_f16(false, a, false, b,
                                                              (short)0, acc2[nt], false, false);
        }
    }

    // ---------------- bias + GELU + dot with W3 + output scatter ----------------
    float part[8];
#pragma unroll
    for (int j = 0; j < 8; ++j) part[j] = 0.f;
#pragma unroll
    for (int nt = 0; nt < 4; ++nt) {
        const int col = nt * 16 + row16;
        const float bb = b2[g * 64 + col];
        const float w3 = W3[g * 64 + col];
#pragma unroll
        for (int j = 0; j < 8; ++j) {
            float x = gelu_exact(acc2[nt][j] + bb);
            part[j] += x * w3;
        }
    }
#pragma unroll
    for (int m = 1; m < 16; m <<= 1)
#pragma unroll
        for (int j = 0; j < 8; ++j) part[j] += __shfl_xor(part[j], m, 32);

    if (row16 == 0) {
        const float bias3 = b3[g];
#pragma unroll
        for (int j = 0; j < 8; ++j) {
            const int r = j + hi16 * 8;
            const int p = gstart + tl * 16 + r;
            if (p < gend) out[p] = part[j] + bias3;
        }
    }
}

// ---------------- launcher ----------------
extern "C" void kernel_launch(void* const* d_in, const int* in_sizes, int n_in,
                              void* d_out, int out_size, void* d_ws, size_t ws_size,
                              hipStream_t stream) {
    const float* emb    = (const float*)d_in[0];
    const int*   edges  = (const int*)  d_in[1];
    const int*   groups = (const int*)  d_in[2];
    const float* W1     = (const float*)d_in[3];
    const float* b1     = (const float*)d_in[4];
    const float* lng    = (const float*)d_in[5];
    const float* lnb    = (const float*)d_in[6];
    const float* W2     = (const float*)d_in[7];
    const float* b2     = (const float*)d_in[8];
    const float* W3     = (const float*)d_in[9];
    const float* b3     = (const float*)d_in[10];

    const int E = in_sizes[2];
    const int G = in_sizes[10];
    const int nblocks = (E + 255) / 256;

    // workspace layout
    char* wsb = (char*)d_ws;
    size_t off = 0;
    int* meta   = (int*)(wsb + off); off += 64;
    int* blkcnt = (int*)(wsb + off); off += (size_t)nblocks * G * 4; off = (off + 63) & ~(size_t)63;
    int* blkoff = (int*)(wsb + off); off += (size_t)nblocks * G * 4; off = (off + 63) & ~(size_t)63;
    int* sorted = (int*)(wsb + off); off += (size_t)E * 4;           off = (off + 63) & ~(size_t)63;
    _Float16* W1t = (_Float16*)(wsb + off); off += (size_t)G * 128 * 256 * 2; off = (off + 63) & ~(size_t)63;
    _Float16* W2t = (_Float16*)(wsb + off); off += (size_t)G * 64 * 128 * 2;

    count_kernel  <<<nblocks, 256, 0, stream>>>(groups, blkcnt, E, G);
    scan_kernel   <<<1, 64, 0, stream>>>(blkcnt, blkoff, meta, nblocks, G);
    scatter_kernel<<<nblocks, 256, 0, stream>>>(groups, blkoff, sorted, E, G);

    const int ntot = G * 128 * 256 + G * 64 * 128;
    prep_weights<<<(ntot + 255) / 256, 256, 0, stream>>>(W1, W2, W1t, W2t, G);

    const int maxTiles = (E + 15) / 16 + G;   // upper bound on padded tile count
    const int mblocks  = (maxTiles + WPB - 1) / WPB;
    const size_t lds_bytes = (size_t)(LDS_HBUF_HALFS + LDS_W1_HALFS + LDS_W2_HALFS) * sizeof(_Float16); // 112KB
    mlp_kernel<<<mblocks, 256, lds_bytes, stream>>>(emb, edges, b1, lng, lnb, b2, W3, b3,
                                                    W1t, W2t, sorted, meta, (float*)d_out, G);
}
